// Head_28492813041878
// MI455X (gfx1250) — compile-verified, matrix operand order checked by reading, not run on hardware
//
#include <hip/hip_runtime.h>

typedef __attribute__((ext_vector_type(16))) __bf16         v16bf;
typedef __attribute__((ext_vector_type(16))) unsigned short v16s;
typedef __attribute__((ext_vector_type(8)))  float          v8f;
typedef __attribute__((ext_vector_type(4)))  float          v4f;
typedef __attribute__((ext_vector_type(2)))  float          v2f;

#define B_     32
#define T_     2048
#define E_     1024
#define H_     64
#define SCALE_ 0.125f   // 64^-0.5

// round-to-nearest-even f32 -> bf16
static __device__ __forceinline__ unsigned short f2bf(float f) {
  unsigned int u = __builtin_bit_cast(unsigned int, f);
  u += 0x7fffu + ((u >> 16) & 1u);
  return (unsigned short)(u >> 16);
}

static __device__ __forceinline__ v8f wmma_bf16(v16bf a, v16bf b, v8f c) {
  return __builtin_amdgcn_wmma_f32_16x16x32_bf16(false, a, false, b, (short)0, c, false, false);
}
static __device__ __forceinline__ v8f wmma_f32(v2f a, v2f b, v8f c) {
  return __builtin_amdgcn_wmma_f32_16x16x4_f32(false, a, false, b, (short)0, c, false, false);
}

// ---------------------------------------------------------------------------
// k0: repack Wq/Wk/Wv [1024x64] f32 -> bf16 in WMMA B-fragment order.
// Fragment id fid = (m*4+n)*32 + s ; per fragment: lane*16 contiguous halves.
// Element e of lane l corresponds to K = s*32 + (l>>4)*8 + (e&7) + (e&8 ? 16:0),
// N = n*16 + (l&15)   (mirrors the 16-bit 16x32 A layout, ISA 7.12.2).
// ---------------------------------------------------------------------------
__global__ void k0_pack_w(const float* __restrict__ Wq, const float* __restrict__ Wk,
                          const float* __restrict__ Wv, unsigned short* __restrict__ Wpk) {
  int tid = blockIdx.x * blockDim.x + threadIdx.x;
  if (tid >= 12288) return;              // 384 fragments * 32 lanes
  int fid  = tid >> 5;
  int lane = tid & 31;
  int m = fid >> 7;
  int n = (fid >> 5) & 3;
  int s = fid & 31;
  const float* W = (m == 0) ? Wq : (m == 1) ? Wk : Wv;
  int nn    = n * 16 + (lane & 15);
  int kbase = s * 32 + (lane >> 4) * 8;
  v16s u;
#pragma unroll
  for (int e = 0; e < 16; ++e) {
    int kk = kbase + (e & 7) + ((e & 8) ? 16 : 0);
    u[e] = f2bf(W[kk * H_ + nn]);
  }
  *(v16s*)(Wpk + (size_t)fid * 512 + lane * 16) = u;
}

// ---------------------------------------------------------------------------
// k1: fused QKV projection (bf16 WMMA) + per-block partial S = K^T V (f32 WMMA)
// grid = 512 blocks (32 batches * 16 chunks of 128 rows), 256 threads = 8 waves,
// one 16-row M-tile per wave.
// ---------------------------------------------------------------------------
__global__ void __launch_bounds__(256)
k1_qkv_ktv(const float* __restrict__ x, const unsigned short* __restrict__ Wpk,
           const float* __restrict__ bq, const float* __restrict__ bk,
           const float* __restrict__ bv, float* __restrict__ Qws,
           float* __restrict__ Spart) {
  __shared__ float kbuf[8192];   // 8 waves * 16 rows * 64 cols
  __shared__ float vbuf[8192];
  const int bat   = blockIdx.x >> 4;
  const int chunk = blockIdx.x & 15;
  const int tid  = threadIdx.x;
  const int wave = tid >> 5, lane = tid & 31;
  const int ln = lane & 15, hi = lane >> 4, mo = hi * 8;
  const int r0 = chunk * 128 + wave * 16;

  // ---- phase 1: Q,K,V tiles; K-dim E=1024 in 32 steps of 32
  const float* xrow = x + ((size_t)(bat * T_ + r0 + ln)) * E_;
  const int kb8 = hi * 8;
  v8f acc[12];                               // [m*4+n] : Q(0..3) K(4..7) V(8..11)
#pragma unroll
  for (int i = 0; i < 12; ++i) acc[i] = (v8f)(0.0f);

  for (int s = 0; s < 32; ++s) {
    const float* ap = xrow + s * 32 + kb8;
    __builtin_prefetch(ap + 32, 0, 3);       // global_prefetch_b8: next k-slab of x
    v4f f0 = *(const v4f*)(ap);
    v4f f1 = *(const v4f*)(ap + 4);
    v4f f2 = *(const v4f*)(ap + 16);
    v4f f3 = *(const v4f*)(ap + 20);
    v16s au;
#pragma unroll
    for (int e = 0; e < 4; ++e) {
      au[e]      = f2bf(f0[e]);
      au[e + 4]  = f2bf(f1[e]);
      au[e + 8]  = f2bf(f2[e]);
      au[e + 12] = f2bf(f3[e]);
    }
    v16bf av = __builtin_bit_cast(v16bf, au);
#pragma unroll
    for (int mn = 0; mn < 12; ++mn) {
      v16s bu = *(const v16s*)(Wpk + ((size_t)(mn * 32 + s)) * 512 + lane * 16);
      acc[mn] = wmma_bf16(av, __builtin_bit_cast(v16bf, bu), acc[mn]);
    }
  }

  // ---- epilogue: add bias; Q -> global ws, K/V -> this wave's LDS region
  {
    size_t qbase = ((size_t)(bat * T_ + r0)) * H_;
    float* kw = kbuf + wave * 1024;
    float* vw = vbuf + wave * 1024;
#pragma unroll
    for (int n = 0; n < 4; ++n) {
      int col = n * 16 + ln;
      float bqv = bq[col], bkv = bk[col], bvv = bv[col];
#pragma unroll
      for (int j = 0; j < 8; ++j) {
        int row = j + mo;                    // C layout: VGPR j -> M = j + 8*hi
        Qws[qbase + (size_t)row * H_ + col] = acc[n][j] + bqv;
        kw[row * 64 + col] = acc[4 + n][j] + bkv;
        vw[row * 64 + col] = acc[8 + n][j] + bvv;
      }
    }
  }

  // ---- phase 2: per-wave partial S = K_tile^T (64x16) @ V_tile (16x64), f32 WMMA
  v8f sacc[16];
#pragma unroll
  for (int i = 0; i < 16; ++i) sacc[i] = (v8f)(0.0f);
  {
    const float* kw = kbuf + wave * 1024;
    const float* vw = vbuf + wave * 1024;
#pragma unroll
    for (int s2 = 0; s2 < 4; ++s2) {
      int t = s2 * 4 + hi * 2;               // f32 A layout: lane holds K = 2*hi, 2*hi+1
      v2f af[4], bf[4];
#pragma unroll
      for (int h = 0; h < 4; ++h) {
        int col = h * 16 + ln;
        af[h].x = kw[t * 64 + col];  af[h].y = kw[(t + 1) * 64 + col];
        bf[h].x = vw[t * 64 + col];  bf[h].y = vw[(t + 1) * 64 + col];
      }
#pragma unroll
      for (int hA = 0; hA < 4; ++hA)
#pragma unroll
        for (int hB = 0; hB < 4; ++hB)
          sacc[hA * 4 + hB] = wmma_f32(af[hA], bf[hB], sacc[hA * 4 + hB]);
    }
  }

  // ---- deterministic barrier-ordered reduction of 8 per-wave partials
  __syncthreads();
  float* sred = kbuf;                        // reuse LDS (phase-2 reads done)
  for (int i = tid; i < 4096; i += 256) sred[i] = 0.0f;
  __syncthreads();
  for (int w = 0; w < 8; ++w) {
    if (wave == w) {
#pragma unroll
      for (int hA = 0; hA < 4; ++hA)
#pragma unroll
        for (int hB = 0; hB < 4; ++hB)
#pragma unroll
          for (int j = 0; j < 8; ++j)
            sred[(hA * 16 + j + mo) * 64 + hB * 16 + ln] += sacc[hA * 4 + hB][j];
    }
    __syncthreads();
  }
  float* outp = Spart + (size_t)blockIdx.x * 4096;
  for (int i = tid; i < 4096; i += 256) outp[i] = sred[i];
}

// ---------------------------------------------------------------------------
// k2: S[b] = SCALE * sum over 16 chunk-partials  (32 blocks x 256 threads)
// ---------------------------------------------------------------------------
__global__ void k2_reduce(const float* __restrict__ Spart, float* __restrict__ S) {
  int b = blockIdx.x, tid = threadIdx.x;
  for (int i = tid; i < 4096; i += 256) {
    float a = 0.0f;
#pragma unroll
    for (int c = 0; c < 16; ++c) a += Spart[((size_t)(b * 16 + c)) * 4096 + i];
    S[(size_t)b * 4096 + i] = a * SCALE_;
  }
}

// ---------------------------------------------------------------------------
// k3: out[b] = Q[b] (2048x64) @ S[b] (64x64), f32 WMMA, S staged in LDS
// ---------------------------------------------------------------------------
__global__ void __launch_bounds__(256)
k3_qs(const float* __restrict__ Qws, const float* __restrict__ S,
      float* __restrict__ out) {
  __shared__ float sbuf[4096];
  const int bat   = blockIdx.x >> 4;
  const int chunk = blockIdx.x & 15;
  const int tid  = threadIdx.x;
  const int wave = tid >> 5, lane = tid & 31;
  const int ln = lane & 15, hi = lane >> 4, mo = hi * 8;
  for (int i = tid; i < 4096; i += 256) sbuf[i] = S[(size_t)bat * 4096 + i];
  __syncthreads();

  const int r0 = chunk * 128 + wave * 16;
  const float* qrow = Qws + ((size_t)(bat * T_ + r0 + ln)) * H_;
  v8f acc[4];
#pragma unroll
  for (int n = 0; n < 4; ++n) acc[n] = (v8f)(0.0f);

#pragma unroll
  for (int s = 0; s < 16; ++s) {
    int k0 = s * 4 + hi * 2;
    v2f a = *(const v2f*)(qrow + k0);        // contiguous K pair per lane
#pragma unroll
    for (int n = 0; n < 4; ++n) {
      v2f bfr;
      bfr.x = sbuf[k0 * 64 + n * 16 + ln];
      bfr.y = sbuf[(k0 + 1) * 64 + n * 16 + ln];
      acc[n] = wmma_f32(a, bfr, acc[n]);
    }
  }

  size_t obase = ((size_t)(bat * T_ + r0)) * H_;
#pragma unroll
  for (int n = 0; n < 4; ++n)
#pragma unroll
    for (int j = 0; j < 8; ++j)
      out[obase + (size_t)(j + mo) * H_ + n * 16 + ln] = acc[n][j];
}

// ---------------------------------------------------------------------------
extern "C" void kernel_launch(void* const* d_in, const int* in_sizes, int n_in,
                              void* d_out, int out_size, void* d_ws, size_t ws_size,
                              hipStream_t stream) {
  const float* x  = (const float*)d_in[0];
  const float* Wq = (const float*)d_in[1];
  const float* bq = (const float*)d_in[2];
  const float* Wk = (const float*)d_in[3];
  const float* bk = (const float*)d_in[4];
  const float* Wv = (const float*)d_in[5];
  const float* bv = (const float*)d_in[6];

  char* ws = (char*)d_ws;
  unsigned short* Wpk  = (unsigned short*)(ws);                       // 384 KB
  float*          Qws  = (float*)(ws + (size_t)(512 * 1024));         // 16 MB
  float*          Spart= (float*)(ws + (size_t)(18 * 1024 * 1024));   // 8 MB
  float*          S    = (float*)(ws + (size_t)(28 * 1024 * 1024));   // 512 KB
  float*          out  = (float*)d_out;

  k0_pack_w <<<48,  256, 0, stream>>>(Wq, Wk, Wv, Wpk);
  k1_qkv_ktv<<<512, 256, 0, stream>>>(x, Wpk, bq, bk, bv, Qws, Spart);
  k2_reduce <<<32,  256, 0, stream>>>(Spart, S);
  k3_qs     <<<512, 256, 0, stream>>>(Qws, S, out);
}
